// Linear4bit_27092653703540
// MI455X (gfx1250) — compile-verified, matrix operand order checked by reading, not run on hardware
//
#include <hip/hip_runtime.h>
#include <hip/hip_bf16.h>
#include <stdint.h>

typedef __attribute__((ext_vector_type(16))) __bf16 v16bf;
typedef __attribute__((ext_vector_type(8)))  __bf16 v8bf;
typedef __attribute__((ext_vector_type(2)))  __bf16 v2bf;
typedef __attribute__((ext_vector_type(8)))  float  v8f;

#define K_DIM   4096
#define N_DIM   4096
#define BM      128
#define BN      256
#define BK      32
#define LDSTR   40   // 32 bf16 + 8 pad -> 80B row stride, conflict-free b128 frag reads

// ---------------------------------------------------------------------------
// Kernel 1: NF4 dequant.  packed[i] holds one byte (two 4-bit codes).
// element 2i = low nibble, element 2i+1 = high nibble; block = (2i)/64 = i/32.
// ---------------------------------------------------------------------------
__global__ void nf4_dequant_kernel(const int* __restrict__ packed,
                                   const float* __restrict__ scales,
                                   const float* __restrict__ table,
                                   __bf16* __restrict__ w,
                                   int n_half) {
    __shared__ float tbl[16];
    if (threadIdx.x < 16) tbl[threadIdx.x] = table[threadIdx.x];
    __syncthreads();

    int i = blockIdx.x * blockDim.x + threadIdx.x;
    if (i >= n_half) return;

    int byte = packed[i];
    float s  = scales[i >> 5];
    float lo = tbl[byte & 15] * s;
    float hi = tbl[(byte >> 4) & 15] * s;

    v2bf o;
    o[0] = (__bf16)lo;
    o[1] = (__bf16)hi;
    *(v2bf*)(&w[(size_t)2 * i]) = o;   // one coalesced 32-bit store
}

// ---------------------------------------------------------------------------
// gfx1250 async copy: global -> LDS, 16B per lane, tracked by ASYNCcnt.
// (inline asm: portable across ROCm 7.2 / amdgpu-toolchain builtin arities)
// ---------------------------------------------------------------------------
__device__ __forceinline__ void async_b128(unsigned lds_off, const void* gptr) {
    asm volatile("global_load_async_to_lds_b128 %0, %1, off"
                 :: "v"(lds_off), "v"(gptr) : "memory");
}
__device__ __forceinline__ void wait_async0() {
    asm volatile("s_wait_asynccnt 0x0" ::: "memory");
}
__device__ __forceinline__ unsigned lds_off32(const void* p) {
    // generic pointer to LDS: low 32 bits are the wave-relative LDS address
    return (unsigned)(uintptr_t)p;
}

// Issue one K-step's tiles: A = BM x BK (512 x 16B chunks), B = BN x BK (1024)
__device__ __forceinline__ void issue_tile(const __bf16* __restrict__ gA,
                                           const __bf16* __restrict__ gB,
                                           __bf16* dA, __bf16* dB, int tid) {
#pragma unroll
    for (int it = 0; it < 2; ++it) {                 // 2 x 256 = 512 chunks
        const int idx = tid + it * 256;
        const int row = idx >> 2, cq = (idx & 3) * 8;
        async_b128(lds_off32(&dA[row * LDSTR + cq]),
                   &gA[(size_t)row * K_DIM + cq]);
    }
#pragma unroll
    for (int it = 0; it < 4; ++it) {                 // 4 x 256 = 1024 chunks
        const int idx = tid + it * 256;
        const int row = idx >> 2, cq = (idx & 3) * 8;
        async_b128(lds_off32(&dB[row * LDSTR + cq]),
                   &gB[(size_t)row * K_DIM + cq]);
    }
}

// ---------------------------------------------------------------------------
// Fragment load: per the CDNA5 16-bit A/B VGPR layout, lane l<16 holds
// K runs {0..7, 16..23}, lane>=16 holds {8..15, 24..31}; two ds_load_b128.
// ---------------------------------------------------------------------------
__device__ __forceinline__ v16bf load_frag(const __bf16* row_base, int ksel) {
    const v8bf lo = *(const v8bf*)(row_base + ksel * 8);
    const v8bf hi = *(const v8bf*)(row_base + ksel * 8 + 16);
    v16bf r;
#pragma unroll
    for (int i = 0; i < 8; ++i) { r[i] = lo[i]; r[i + 8] = hi[i]; }
    return r;
}

// ---------------------------------------------------------------------------
// Kernel 2: bf16 WMMA GEMM.  C[M,N] = X[M,K] * W[N,K]^T.
// 256 threads = 8 waves (2x4); wave tile 64x64 = 4x4 wmma accumulators.
// Double-buffered LDS fed by async-to-LDS; 1 barrier + 1 asynccnt wait / step.
// ---------------------------------------------------------------------------
__global__ void __launch_bounds__(256)
nf4_gemm_kernel(const __bf16* __restrict__ x,
                const __bf16* __restrict__ w,
                __bf16* __restrict__ out) {
    __shared__ __align__(16) __bf16 sA[2][BM * LDSTR];   // 2 x 10240 B
    __shared__ __align__(16) __bf16 sB[2][BN * LDSTR];   // 2 x 20480 B

    const int tid   = threadIdx.x;
    const int lane  = tid & 31;
    const int wave  = tid >> 5;
    const int waveM = wave >> 2;        // 0..1 -> M offset *64
    const int waveN = wave & 3;         // 0..3 -> N offset *64
    const int ksel  = lane >> 4;        // which K runs this lane owns
    const int l16   = lane & 15;

    const int m0 = blockIdx.y * BM;
    const int n0 = blockIdx.x * BN;
    const __bf16* gA = x + (size_t)m0 * K_DIM;
    const __bf16* gB = w + (size_t)n0 * K_DIM;

    v8f acc[4][4];
#pragma unroll
    for (int mi = 0; mi < 4; ++mi)
#pragma unroll
        for (int ni = 0; ni < 4; ++ni)
            acc[mi][ni] = (v8f){0.f, 0.f, 0.f, 0.f, 0.f, 0.f, 0.f, 0.f};

    // prologue: stage 0 in flight
    issue_tile(gA, gB, sA[0], sB[0], tid);

    const int NK = K_DIM / BK;   // 128
    for (int i = 0; i < NK; ++i) {
        const int cur = i & 1;

        wait_async0();           // my stage-cur chunks have landed in LDS
        __syncthreads();         // everyone's landed; prev readers of next done

        if (i + 1 < NK)
            issue_tile(gA + (i + 1) * BK, gB + (i + 1) * BK,
                       sA[cur ^ 1], sB[cur ^ 1], tid);
        if (i + 4 < NK) {        // warm L2 four K-steps ahead
            __builtin_prefetch(&gA[(size_t)(tid >> 1) * K_DIM + (i + 4) * BK], 0, 0);
            __builtin_prefetch(&gB[(size_t)tid * K_DIM + (i + 4) * BK], 0, 0);
        }

        const __bf16* cA = sA[cur];
        const __bf16* cB = sB[cur];

        v16bf afr[4], bfr[4];
#pragma unroll
        for (int mi = 0; mi < 4; ++mi) {
            const int row = waveM * 64 + mi * 16 + l16;
            afr[mi] = load_frag(&cA[row * LDSTR], ksel);
        }
#pragma unroll
        for (int ni = 0; ni < 4; ++ni) {
            const int col = waveN * 64 + ni * 16 + l16;
            bfr[ni] = load_frag(&cB[col * LDSTR], ksel);
        }

#pragma unroll
        for (int mi = 0; mi < 4; ++mi)
#pragma unroll
            for (int ni = 0; ni < 4; ++ni)
                acc[mi][ni] = __builtin_amdgcn_wmma_f32_16x16x32_bf16(
                    false, afr[mi], false, bfr[ni],
                    (short)0, acc[mi][ni], false, false);
    }

    // Epilogue: C/D layout -> VGPR r: lanes 0-15 row r, lanes 16-31 row r+8.
    const int rowAdd = (lane >= 16) ? 8 : 0;
#pragma unroll
    for (int mi = 0; mi < 4; ++mi) {
#pragma unroll
        for (int ni = 0; ni < 4; ++ni) {
            const int cbase = n0 + waveN * 64 + ni * 16 + l16;
            const int rbase = m0 + waveM * 64 + mi * 16 + rowAdd;
#pragma unroll
            for (int r = 0; r < 8; ++r)
                out[(size_t)(rbase + r) * N_DIM + cbase] = (__bf16)acc[mi][ni][r];
        }
    }
}

extern "C" void kernel_launch(void* const* d_in, const int* in_sizes, int n_in,
                              void* d_out, int out_size, void* d_ws, size_t ws_size,
                              hipStream_t stream) {
    const __bf16* x      = (const __bf16*)d_in[0];
    const int*    packed = (const int*)d_in[1];
    const float*  scales = (const float*)d_in[2];
    const float*  table  = (const float*)d_in[3];

    __bf16* w   = (__bf16*)d_ws;           // 4096*4096 bf16 = 32 MB scratch
    __bf16* out = (__bf16*)d_out;

    const int n_half = in_sizes[1];        // OUT_F*IN_F/2 packed bytes
    nf4_dequant_kernel<<<(n_half + 255) / 256, 256, 0, stream>>>(
        packed, scales, table, w, n_half);

    const int M = in_sizes[0] / K_DIM;     // tokens (8192)
    dim3 grid(N_DIM / BN, M / BM);
    nf4_gemm_kernel<<<grid, 256, 0, stream>>>(x, w, out);
}
